// StressLSTM_72791105733184
// MI455X (gfx1250) — matrix operand order repaired and checked by reading, not compiled, not run on hardware
//
#include <hip/hip_runtime.h>

// CDNA5 / gfx1250 LSTM forward (B=8192, T=1024, D_IN=4, H=64) + MLP head.
// 256 workgroups x 32 batch rows; 8 waves/WG; each wave owns a
// (row-half, j-block) 16x16 patch and computes all 4 gate tiles for it so
// the cell update is wave-local. Recurrence: v_wmma_f32_16x16x32_f16
// (f16 in, f32 accumulate); K=4 input projection: v_wmma_f32_16x16x4_f32.
// Nonlinearities use gfx1250's native v_tanh_f32; the 0.5x sigmoid input
// scale is pre-folded into the i/f/o gate weights (exact in f16/f32).

typedef float    v2f  __attribute__((ext_vector_type(2)));
typedef float    v4f  __attribute__((ext_vector_type(4)));
typedef float    v8f  __attribute__((ext_vector_type(8)));
typedef _Float16 v8h  __attribute__((ext_vector_type(8)));
typedef _Float16 v16h __attribute__((ext_vector_type(16)));

#define D_IN 4
#define HDIM 64
#define G4   256      // 4*H
#define FC1N 32
#define NCLS 2
#define TLEN 1024
#define BTOT 8192
#define ROWS 32       // batch rows per workgroup
#define TC   32       // timestep chunk staged in LDS
#define NTHR 256      // 8 waves

#if __has_builtin(__builtin_amdgcn_tanhf)
__device__ __forceinline__ float tanh_f(float x) { return __builtin_amdgcn_tanhf(x); }
// Input already pre-scaled by 0.5 via the weights: sigmoid(2y) = 0.5 + 0.5*tanh(y)
__device__ __forceinline__ float sigm_pre(float y) {
    return __builtin_fmaf(0.5f, __builtin_amdgcn_tanhf(y), 0.5f);
}
#else
__device__ __forceinline__ float tanh_f(float x) {
    return __builtin_fmaf(2.0f, __builtin_amdgcn_rcpf(1.0f + __expf(-2.0f * x)), -1.0f);
}
__device__ __forceinline__ float sigm_pre(float y) {   // sigmoid(2y)
    return __builtin_amdgcn_rcpf(1.0f + __expf(-2.0f * y));
}
#endif

__global__ __launch_bounds__(NTHR)
void lstm_head_kernel(const float* __restrict__ x,
                      const float* __restrict__ Wih,  const float* __restrict__ Whh,
                      const float* __restrict__ bih,  const float* __restrict__ bhh,
                      const float* __restrict__ Wfc1, const float* __restrict__ bfc1,
                      const float* __restrict__ Wfc2, const float* __restrict__ bfc2,
                      float* __restrict__ out)
{
    __shared__ _Float16 whh_lds[G4 * HDIM];       // 32 KB, W_hh as f16 (i/f/o pre-halved)
    __shared__ _Float16 h_lds[ROWS * HDIM];       //  4 KB, h_t as f16
    __shared__ float    xbuf[TC * ROWS * D_IN];   // 16 KB, staged x chunk [s][r][d]
    __shared__ float    hbuf[ROWS * HDIM];        //  8 KB, final h (f32)
    __shared__ float    zbuf[ROWS * FC1N];        //  4 KB, fc1 activations

    const int tid   = threadIdx.x;
    const int lane  = tid & 31;
    const int wave  = tid >> 5;
    const int rh    = wave >> 2;       // 0..1 : which 16-row half
    const int jb    = wave & 3;        // 0..3 : which 16-col block of H
    const int jbase = jb * 16;
    const int ln    = lane & 15;
    const int hi    = lane >> 4;       // lane-half
    const int rbase = blockIdx.x * ROWS;

    // ---- stage weights / init state ----
    // Gates i(0), f(1), o(3) are pre-scaled by 0.5 (sigmoid fold); g(2) is not.
    for (int i = tid; i < G4 * HDIM; i += NTHR) {
        const float sc = ((i >> 12) == 2) ? 1.0f : 0.5f;   // i>>12 == gate index
        whh_lds[i] = (_Float16)(Whh[i] * sc);
    }
    for (int i = tid; i < ROWS * HDIM; i += NTHR) h_lds[i] = (_Float16)0.0f;
    __syncthreads();

    // Per-lane loop-invariant fragments.
    const int k0 = hi ? 2 : 0;         // f32 A/B K-base (16x16x4 layout)
    float biasv[4];
    v2f   wihf[4];                     // W_ih^T B-frags (lane=N, VGPR=K)
    v16h  bfrag[4][2];                 // W_hh^T B-frags: [gate][K-half]
    #pragma unroll
    for (int g = 0; g < 4; ++g) {
        const float sc = (g == 2) ? 1.0f : 0.5f;
        const int nglob = g * HDIM + jbase + ln;     // global gate index (N)
        biasv[g] = (bih[nglob] + bhh[nglob]) * sc;
        v2f w    = *(const v2f*)(Wih + nglob * D_IN + k0);
        wihf[g]  = w * sc;
        #pragma unroll
        for (int kh = 0; kh < 2; ++kh) {
            // B 16-bit layout: lane=N, lanes0-15 K {kb..kb+7, kb+16..kb+23}.
            const int kb = kh * 32 + (hi ? 8 : 0);
            const _Float16* p = whh_lds + nglob * HDIM + kb;
            union { v16h v; v8h h8[2]; } u;
            u.h8[0] = *(const v8h*)(p);
            u.h8[1] = *(const v8h*)(p + 16);
            bfrag[g][kh] = u.v;
        }
    }

    v8f cstate = {0.f, 0.f, 0.f, 0.f, 0.f, 0.f, 0.f, 0.f};
    v8f hlast  = {0.f, 0.f, 0.f, 0.f, 0.f, 0.f, 0.f, 0.f};

    // ---- time loop ----
    for (int t0 = 0; t0 < TLEN; t0 += TC) {
        // Stage x[:, t0:t0+TC, :] for our 32 rows (coalesced 512B runs).
        #pragma unroll
        for (int i = 0; i < (TC * ROWS) / NTHR; ++i) {
            const int idx = tid + i * NTHR;          // 0..1023
            const int r = idx >> 5, s = idx & 31;
            const size_t gofs = ((size_t)(rbase + r) * TLEN + t0 + s) * D_IN;
            *(v4f*)(xbuf + (s * ROWS + r) * D_IN) = *(const v4f*)(x + gofs);
            if (t0 + TC < TLEN)                       // prefetch next chunk
                __builtin_prefetch(x + gofs + (size_t)TC * D_IN, 0, 0);
        }
        __syncthreads();

        for (int s = 0; s < TC; ++s) {
            // A-frags of h (16-bit A layout: lanes0-15 K {kb..kb+7, kb+16..kb+23}).
            const int arow = rh * 16 + ln;
            v16h afrag[2];
            #pragma unroll
            for (int kh = 0; kh < 2; ++kh) {
                const int kb = kh * 32 + (hi ? 8 : 0);
                const _Float16* p = h_lds + arow * HDIM + kb;
                union { v16h v; v8h h8[2]; } u;
                u.h8[0] = *(const v8h*)(p);
                u.h8[1] = *(const v8h*)(p + 16);
                afrag[kh] = u.v;
            }
            // x A-frag (f32 16x4: lanes0-15 K{0,1}, lanes16-31 K{2,3}).
            const v2f xf = *(const v2f*)(xbuf + (s * ROWS + arow) * D_IN + k0);

            v8f d[4];
            #pragma unroll
            for (int g = 0; g < 4; ++g) {
                const float b = biasv[g];
                v8f cc = {b, b, b, b, b, b, b, b};
                cc = __builtin_amdgcn_wmma_f32_16x16x4_f32(
                         false, xf, false, wihf[g], (short)0, cc, false, false);
                cc = __builtin_amdgcn_wmma_f32_16x16x32_f16(
                         false, afrag[0], false, bfrag[g][0], (short)0, cc, false, false);
                cc = __builtin_amdgcn_wmma_f32_16x16x32_f16(
                         false, afrag[1], false, bfrag[g][1], (short)0, cc, false, false);
                d[g] = cc;
            }

            // Wave-local cell update: lane owns (m, j) for all four gates.
            #pragma unroll
            for (int v = 0; v < 8; ++v) {
                const float iv = sigm_pre(d[0][v]);   // weights pre-scaled 0.5
                const float fv = sigm_pre(d[1][v]);
                const float gv = tanh_f(d[2][v]);
                const float ov = sigm_pre(d[3][v]);
                const float cv = __builtin_fmaf(fv, cstate[v], iv * gv);
                cstate[v] = cv;
                const float hv = ov * tanh_f(cv);
                hlast[v] = hv;                         // register only; no branch
                const int mrow = rh * 16 + v + (hi ? 8 : 0);
                h_lds[mrow * HDIM + jbase + ln] = (_Float16)hv;
            }
            __syncthreads();   // h_t complete before anyone reads it / restages x
        }
    }

    // ---- dump final h (f32) once, then MLP head ----
    #pragma unroll
    for (int v = 0; v < 8; ++v) {
        const int mrow = rh * 16 + v + (hi ? 8 : 0);
        hbuf[mrow * HDIM + jbase + ln] = hlast[v];
    }
    __syncthreads();

    // relu(h @ Wfc1^T + b1) @ Wfc2^T + b2
    for (int idx = tid; idx < ROWS * FC1N; idx += NTHR) {
        const int m = idx >> 5, kk = idx & 31;
        float acc = bfc1[kk];
        #pragma unroll 8
        for (int j = 0; j < HDIM; ++j) acc += hbuf[m * HDIM + j] * Wfc1[kk * HDIM + j];
        zbuf[m * FC1N + kk] = fmaxf(acc, 0.0f);
    }
    __syncthreads();
    if (tid < ROWS * NCLS) {
        const int m = tid >> 1, cls = tid & 1;
        float acc = bfc2[cls];
        #pragma unroll
        for (int kk = 0; kk < FC1N; ++kk) acc += zbuf[m * FC1N + kk] * Wfc2[cls * FC1N + kk];
        out[(size_t)(rbase + m) * NCLS + cls] = acc;
    }
}

extern "C" void kernel_launch(void* const* d_in, const int* in_sizes, int n_in,
                              void* d_out, int out_size, void* d_ws, size_t ws_size,
                              hipStream_t stream) {
    (void)in_sizes; (void)n_in; (void)out_size; (void)d_ws; (void)ws_size;
    const float* x    = (const float*)d_in[0];
    const float* Wih  = (const float*)d_in[1];
    const float* Whh  = (const float*)d_in[2];
    const float* bih  = (const float*)d_in[3];
    const float* bhh  = (const float*)d_in[4];
    const float* Wfc1 = (const float*)d_in[5];
    const float* bfc1 = (const float*)d_in[6];
    const float* Wfc2 = (const float*)d_in[7];
    const float* bfc2 = (const float*)d_in[8];
    float* out = (float*)d_out;

    dim3 grid(BTOT / ROWS), block(NTHR);
    lstm_head_kernel<<<grid, block, 0, stream>>>(x, Wih, Whh, bih, bhh,
                                                 Wfc1, bfc1, Wfc2, bfc2, out);
}
